// SoftmaxRefMatcher_9457517986536
// MI455X (gfx1250) — matrix operand order repaired and checked by reading, not compiled, not run on hardware
//
#include <hip/hip_runtime.h>
#include <hip/hip_bf16.h>

typedef __attribute__((ext_vector_type(16))) _Float16 v16h;
typedef __attribute__((ext_vector_type(8)))  float    v8f;

#define NPAIR   2
#define CC      64
#define NPTS    400
#define NTILES  25          // 400 / 16
#define IMGW    256
#define HW      65536       // 256*256
#define MTILES  4096        // HW / 16
#define NCHUNK  64
#define MT_PER_CHUNK 64     // (HW/16)/NCHUNK -> 1024 columns per chunk
#define TEMP_INV 100.0f

// workspace layout (halves / bytes)
static constexpr size_t SRC_HALVES = (size_t)NPAIR * MTILES * 2 * 512;   // 8,388,608 halves = 16 MB
static constexpr size_t TGT_HALVES = (size_t)NPAIR * NTILES * 2 * 512;   // 51,200 halves
static constexpr size_t PART_OFF_BYTES = (SRC_HALVES + TGT_HALVES) * 2;  // 16,879,616 (16B aligned)

// ---------------------------------------------------------------------------
// Normalize desc_dense columns over C and pack into WMMA B-fragment layout.
// B 32x16 f16 packing used here (contiguous k, same as A): lanes 0-15 hold N=lane, halves h -> K=h (K 0..15);
// lanes 16-31 hold N=lane-16, halves h -> K=16+h. Two frags cover K=0..63.
// ---------------------------------------------------------------------------
__global__ void prep_src(const float* __restrict__ desc_dense,
                         const int* __restrict__ src_ids,
                         _Float16* __restrict__ srcfrag) {
    int tid = blockIdx.x * blockDim.x + threadIdx.x;
    if (tid >= NPAIR * HW) return;
    int p = tid / HW, m = tid % HW;
    int sb = src_ids[p];
    const float* base = desc_dense + (size_t)sb * CC * HW + m;
    float vals[CC];
    float ss = 0.f;
#pragma unroll
    for (int c = 0; c < CC; ++c) { float x = base[(size_t)c * HW]; vals[c] = x; ss += x * x; }
    float inv = 1.0f / fmaxf(sqrtf(ss), 1e-12f);
    int mtile = m >> 4, nl = m & 15;
    _Float16* outb = srcfrag + (size_t)(p * MTILES + mtile) * 2 * 512;
#pragma unroll
    for (int k = 0; k < CC; ++k) {
        int f  = k >> 5;
        int kk = k & 31;
        int hi = kk >> 4;
        int h  = kk & 15;
        outb[(size_t)f * 512 + (nl + 16 * hi) * 16 + h] = (_Float16)(vals[k] * inv);
    }
}

// ---------------------------------------------------------------------------
// Normalize keypoint descriptors and pack into WMMA A-fragment layout.
// A 16x32 f16 packing used here (contiguous k, same as B): lane L -> M=L%16, half h -> K=h+16*(L>=16).
// Inverse used here: k -> hi=k>>4, h=k&15, lane=r+16*hi.
// Also emits tgt_scores into the output buffer.
// ---------------------------------------------------------------------------
__global__ void prep_tgt(const float* __restrict__ kp_desc,
                         const float* __restrict__ kp_scores,
                         const int* __restrict__ tgt_ids,
                         _Float16* __restrict__ tgtfrag,
                         float* __restrict__ out) {
    int tid = blockIdx.x * blockDim.x + threadIdx.x;
    if (tid >= NPAIR * NPTS) return;
    int p = tid / NPTS, n = tid % NPTS;
    int tb = tgt_ids[p];
    const float* base = kp_desc + (size_t)tb * CC * NPTS + n;
    float vals[CC];
    float ss = 0.f;
#pragma unroll
    for (int c = 0; c < CC; ++c) { float x = base[(size_t)c * NPTS]; vals[c] = x; ss += x * x; }
    float inv = 1.0f / fmaxf(sqrtf(ss), 1e-12f);
    int ntile = n >> 4, r = n & 15;
    _Float16* outb = tgtfrag + (size_t)(p * NTILES + ntile) * 2 * 512;
#pragma unroll
    for (int k = 0; k < CC; ++k) {
        int f  = k >> 5;
        int kk = k & 31;
        int hi = kk >> 4;
        int h  = kk & 15;
        outb[(size_t)f * 512 + (r + 16 * hi) * 16 + h] = (_Float16)(vals[k] * inv);
    }
    // tgt_scores at offset 2*NPAIR*NPTS (after pseudo_coords)
    out[2 * NPAIR * NPTS + p * NPTS + n] = kp_scores[(size_t)tb * NPTS + n];
}

// ---------------------------------------------------------------------------
// Fused GEMM (WMMA f16->f32) + two-pass softmax + coordinate expectation.
// One wave per (pair, ntile, chunk). C/D layout: lane L holds col N=L%16,
// rows M = r + 8*(L/16) for r=0..7.
// Pass 1: WMMA sweep tracking per-row max (1 VALU/elem).
// Pass 2: WMMA sweep accumulating exp(score-max) sums (1 exp + FMAs/elem).
// WMMAs are cheap and B-fragments are L2-resident, so doubling the sweeps
// to eliminate online-softmax rescaling halves the bound resource (VALU/TRANS).
// ---------------------------------------------------------------------------
__global__ void __launch_bounds__(256)
matmul_softmax(const _Float16* __restrict__ srcfrag,
               const _Float16* __restrict__ tgtfrag,
               float4* __restrict__ partials) {
    int bid   = blockIdx.x;                 // NPAIR * NTILES * 8 blocks
    int p     = bid / (NTILES * 8);
    int rem   = bid % (NTILES * 8);
    int ntile = rem >> 3;
    int cg    = rem & 7;
    int wave  = threadIdx.x >> 5;
    int lane  = threadIdx.x & 31;
    int chunk = cg * 8 + wave;
    int hi    = lane >> 4;

    const _Float16* ab = tgtfrag + (size_t)(p * NTILES + ntile) * 2 * 512 + lane * 16;
    const v16h a0 = *(const v16h*)(ab);
    const v16h a1 = *(const v16h*)(ab + 512);

    const _Float16* chunkbase =
        srcfrag + (size_t)(p * MTILES + chunk * MT_PER_CHUNK) * 2 * 512 + lane * 16;

    // ---------------- pass 1: per-row maximum of raw dot products ----------
    float rmax[8];
#pragma unroll
    for (int r = 0; r < 8; ++r) rmax[r] = -1e30f;

#pragma unroll 2
    for (int t = 0; t < MT_PER_CHUNK; ++t) {
        const _Float16* sb = chunkbase + (size_t)t * 1024;
        v16h b0 = *(const v16h*)(sb);
        v16h b1 = *(const v16h*)(sb + 512);
        v8f acc = {};
        acc = __builtin_amdgcn_wmma_f32_16x16x32_f16(false, a0, false, b0, (short)0, acc, false, false);
        acc = __builtin_amdgcn_wmma_f32_16x16x32_f16(false, a1, false, b1, (short)0, acc, false, false);
#pragma unroll
        for (int r = 0; r < 8; ++r) rmax[r] = fmaxf(rmax[r], acc[r]);
    }
    // make the row max uniform across each 16-lane column group
#pragma unroll
    for (int mask = 1; mask < 16; mask <<= 1) {
#pragma unroll
        for (int r = 0; r < 8; ++r)
            rmax[r] = fmaxf(rmax[r], __shfl_xor(rmax[r], mask, 32));
    }
    float xm[8];        // scaled row max (score space)
#pragma unroll
    for (int r = 0; r < 8; ++r) xm[r] = rmax[r] * TEMP_INV;

    // ---------------- pass 2: exp-weighted sums ----------------------------
    float srow[8], urow[8], vrow[8];
#pragma unroll
    for (int r = 0; r < 8; ++r) { srow[r] = 0.f; urow[r] = 0.f; vrow[r] = 0.f; }

#pragma unroll 2
    for (int t = 0; t < MT_PER_CHUNK; ++t) {
        int mtile = chunk * MT_PER_CHUNK + t;
        const _Float16* sb = chunkbase + (size_t)t * 1024;
        v16h b0 = *(const v16h*)(sb);
        v16h b1 = *(const v16h*)(sb + 512);
        v8f acc = {};
        acc = __builtin_amdgcn_wmma_f32_16x16x32_f16(false, a0, false, b0, (short)0, acc, false, false);
        acc = __builtin_amdgcn_wmma_f32_16x16x32_f16(false, a1, false, b1, (short)0, acc, false, false);

        int   m  = (mtile << 4) + (lane & 15);
        float uu = (float)(m & (IMGW - 1));
        float vv = (float)(m >> 8);
#pragma unroll
        for (int r = 0; r < 8; ++r) {
            float e = __expf(fmaf(acc[r], TEMP_INV, -xm[r]));  // <= 0 argument
            srow[r] += e;
            urow[r] = fmaf(e, uu, urow[r]);
            vrow[r] = fmaf(e, vv, vrow[r]);
        }
    }

    // pure-sum reduction across the 16 lanes of each column group
#pragma unroll
    for (int mask = 1; mask < 16; mask <<= 1) {
#pragma unroll
        for (int r = 0; r < 8; ++r) {
            srow[r] += __shfl_xor(srow[r], mask, 32);
            urow[r] += __shfl_xor(urow[r], mask, 32);
            vrow[r] += __shfl_xor(vrow[r], mask, 32);
        }
    }

    if ((lane & 15) == 0) {
#pragma unroll
        for (int r = 0; r < 8; ++r) {
            int row = p * NPTS + ntile * 16 + r + 8 * hi;   // global (pair, keypoint) row
            partials[(size_t)row * NCHUNK + chunk] = make_float4(xm[r], srow[r], urow[r], vrow[r]);
        }
    }
}

// ---------------------------------------------------------------------------
// Merge chunk partials per row, normalize, write coords + ids.
// ---------------------------------------------------------------------------
__global__ void finalize(const float4* __restrict__ partials,
                         const int* __restrict__ tgt_ids,
                         const int* __restrict__ src_ids,
                         float* __restrict__ out) {
    int tid = blockIdx.x * blockDim.x + threadIdx.x;
    if (tid >= NPAIR * NPTS) return;
    float mm = -1e30f, s = 0.f, u = 0.f, v = 0.f;
    const float4* pp = partials + (size_t)tid * NCHUNK;
    for (int c = 0; c < NCHUNK; ++c) {
        float4 q  = pp[c];
        float nm  = fmaxf(mm, q.x);
        float e1  = __expf(mm - nm);
        float e2  = __expf(q.x - nm);
        s = s * e1 + q.y * e2;
        u = u * e1 + q.z * e2;
        v = v * e1 + q.w * e2;
        mm = nm;
    }
    float inv = 1.0f / s;
    out[tid * 2 + 0] = u * inv;
    out[tid * 2 + 1] = v * inv;
    if (tid < NPAIR) {
        out[3 * NPAIR * NPTS + tid]         = (float)tgt_ids[tid];   // 2400, 2401
        out[3 * NPAIR * NPTS + NPAIR + tid] = (float)src_ids[tid];   // 2402, 2403
    }
}

extern "C" void kernel_launch(void* const* d_in, const int* in_sizes, int n_in,
                              void* d_out, int out_size, void* d_ws, size_t ws_size,
                              hipStream_t stream) {
    const float* kp_scores  = (const float*)d_in[0];   // [4,1,400]
    const float* kp_desc    = (const float*)d_in[1];   // [4,64,400]
    const float* desc_dense = (const float*)d_in[2];   // [4,64,256,256]
    const int*   tgt_ids    = (const int*)d_in[3];     // [2]
    const int*   src_ids    = (const int*)d_in[4];     // [2]
    float* out = (float*)d_out;                        // 2404 floats

    _Float16* srcfrag = (_Float16*)d_ws;
    _Float16* tgtfrag = srcfrag + SRC_HALVES;
    float4*   partials = (float4*)((char*)d_ws + PART_OFF_BYTES);

    prep_src<<<(NPAIR * HW + 255) / 256, 256, 0, stream>>>(desc_dense, src_ids, srcfrag);
    prep_tgt<<<(NPAIR * NPTS + 255) / 256, 256, 0, stream>>>(kp_desc, kp_scores, tgt_ids, tgtfrag, out);
    matmul_softmax<<<NPAIR * NTILES * 8, 256, 0, stream>>>(srcfrag, tgtfrag, partials);
    finalize<<<(NPAIR * NPTS + 255) / 256, 256, 0, stream>>>(partials, tgt_ids, src_ids, out);
}
